// Net_79121887527491
// MI455X (gfx1250) — compile-verified, matrix operand order checked by reading, not run on hardware
//
#include <hip/hip_runtime.h>

// LSTM (H=32, B=256, T=2048) + FC(32->2) for MI455X / gfx1250.
//
//  - Kernel 1 (parallel): xg[t,b,4H] = emb[x[b,t]] @ W_ih + b via
//    v_wmma_f32_16x16x32_f16 (K=32 == H exactly), stored f32 in d_ws in the
//    scan kernel's accumulator layout (T*B*4H*4 = 256 MiB of ws).
//  - Kernel 2 (serial scan): 16 independent chains (one wave per 16-batch
//    tile). xg is staged through a 4-deep LDS ring filled with
//    global_load_async_to_lds_b128 issued 3 steps ahead (s_wait_asynccnt),
//    so the serial loop sees LDS latency instead of HBM latency.
//    Per step: 8 WMMAs for h @ W_hh, lane-local gate math via native
//    v_tanh_f32, h repacked directly into the next step's B operand
//    (gate/hidden mapping co-designed with the WMMA lane-half K split).

typedef __attribute__((ext_vector_type(16))) _Float16 v16h;
typedef __attribute__((ext_vector_type(8)))  float    v8f;
typedef __attribute__((ext_vector_type(4)))  float    v4f;

#define HSZ    32
#define G4     128          // 4*H gate width
#define BSZ    256
#define TSZ    2048
#define NT     8            // gate tiles of 16 per step
#define TCHUNK 16           // timesteps per block in kernel 1

#define NSTAGE 4            // LDS ring depth (power of 2)
#define LA     3            // async prefetch distance (16 loads/stage -> wait 48)

#define LOG2E  1.4426950408889634f

__device__ __forceinline__ float fast_tanh(float x) {
#if __has_builtin(__builtin_amdgcn_tanhf)
    return __builtin_amdgcn_tanhf(x);          // v_tanh_f32 (gfx1250 TRANS op)
#else
    float e = __builtin_amdgcn_exp2f(2.0f * LOG2E * x);
    return 1.0f - 2.0f * __builtin_amdgcn_rcpf(e + 1.0f);
#endif
}
__device__ __forceinline__ float fast_sigmoid(float x) {
    // sigmoid(x) = 0.5 + 0.5*tanh(x/2): one TRANS op, no reciprocal
    return __builtin_fmaf(0.5f, fast_tanh(0.5f * x), 0.5f);
}

// Gate-column mapping: tile t (0..7), D-row m (0..15) -> column of W / index
// of bias.  gate = t>>1, hidden unit u = 16*(t&1) + 8*(m>>3) + (m&7).
// Keeps all four gates of hidden unit u in one lane half and makes the
// produced h land in B-operand K slot K(hi,s) = 16*(s>>3) + 8*hi + (s&7).
__device__ __forceinline__ int gate_col(int t, int m) {
    return 32 * (t >> 1) + 16 * (t & 1) + 8 * (m >> 3) + (m & 7);
}

// Load W^T (W is [32 x 128] row-major) into 8 A-matrix tiles (16x32 f16).
__device__ __forceinline__ void load_w_tiles(const float* __restrict__ W,
                                             v16h* A, int lane) {
    const int m  = lane & 15;
    const int hi = lane >> 4;
#pragma unroll
    for (int t = 0; t < NT; ++t) {
        const int col = gate_col(t, m);
        v16h a;
#pragma unroll
        for (int s = 0; s < 16; ++s) {
            const int row = 16 * (s >> 3) + 8 * hi + (s & 7);
            a[s] = (_Float16)W[row * G4 + col];
        }
        A[t] = a;
    }
}

__device__ __forceinline__ size_t xg_base(int t, int btile, int lane) {
    // [t][btile][tile][lane][8 floats]; tile stride = 32*8 = 256 floats
    return ((((size_t)t * 16 + btile) * NT) * 32 + lane) * 8;
}

// ---------------- Kernel 1: xg = emb[x] @ W_ih + b (parallel) ----------------
__global__ __launch_bounds__(32, 1)
void lstm_xgate_kernel(const int* __restrict__ x,
                       const float* __restrict__ emb,
                       const float* __restrict__ W_ih,
                       const float* __restrict__ bias,
                       float* __restrict__ xg) {
    const int lane  = threadIdx.x;
    const int btile = blockIdx.x;            // 0..15
    const int t0    = blockIdx.y * TCHUNK;   // timestep chunk
    const int m     = lane & 15;
    const int hi    = lane >> 4;

    v16h A[NT];
    load_w_tiles(W_ih, A, lane);

    // Bias in accumulator (D) layout: VGPR r, lane half hi -> row m = r + 8*hi
    v8f bia[NT];
#pragma unroll
    for (int t8 = 0; t8 < NT; ++t8)
#pragma unroll
        for (int r = 0; r < 8; ++r)
            bia[t8][r] = bias[gate_col(t8, r + 8 * hi)];

    const int b = btile * 16 + m;
    const int* __restrict__ xrow = x + (size_t)b * TSZ;

    for (int tt = 0; tt < TCHUNK; ++tt) {
        const int t = t0 + tt;
        const float* __restrict__ erow = emb + (size_t)xrow[t] * HSZ;

        // B operand = e^T; lane-half K split matches the A layout.
        v4f e0 = *(const v4f*)(erow + 8 * hi);
        v4f e1 = *(const v4f*)(erow + 8 * hi + 4);
        v4f e2 = *(const v4f*)(erow + 16 + 8 * hi);
        v4f e3 = *(const v4f*)(erow + 16 + 8 * hi + 4);
        v16h Bv;
#pragma unroll
        for (int q = 0; q < 4; ++q) {
            Bv[q]      = (_Float16)e0[q];
            Bv[4 + q]  = (_Float16)e1[q];
            Bv[8 + q]  = (_Float16)e2[q];
            Bv[12 + q] = (_Float16)e3[q];
        }

        float* __restrict__ base = xg + xg_base(t, btile, lane);
#pragma unroll
        for (int t8 = 0; t8 < NT; ++t8) {
            v8f acc = __builtin_amdgcn_wmma_f32_16x16x32_f16(
                false, A[t8], false, Bv, (short)0, bia[t8], false, false);
            *(v8f*)(base + (size_t)t8 * 256) = acc;
        }
    }
}

// ---------------- Kernel 2: serial scan over T + final FC ----------------
__global__ __launch_bounds__(32, 1)
void lstm_scan_kernel(const float* __restrict__ W_hh,
                      const float* __restrict__ xg,
                      const float* __restrict__ fc_w,
                      const float* __restrict__ fc_b,
                      float* __restrict__ out) {
    // 4-deep ring of xg stages: [stage][tile][lane][8 floats] = 32 KB LDS.
    __shared__ __align__(16) float xstage[NSTAGE][NT][32][8];

    const int lane  = threadIdx.x;
    const int btile = blockIdx.x;            // 0..15 -> 16 independent chains
    const int m     = lane & 15;
    const int hi    = lane >> 4;

    v16h A[NT];
    load_w_tiles(W_hh, A, lane);

    // Issue one stage (16 async b128s: 8 tiles x 32B per lane) LA steps ahead.
    auto issue_stage = [&](int t, int slot) {
        const float* __restrict__ gsrc = xg + xg_base(t, btile, lane);
#pragma unroll
        for (int t8 = 0; t8 < NT; ++t8) {
            unsigned loff = (unsigned)(uintptr_t)&xstage[slot][t8][lane][0];
            unsigned long long ga =
                (unsigned long long)(uintptr_t)(gsrc + (size_t)t8 * 256);
            // INST_OFFSET applies to both the LDS and global addresses.
            asm volatile(
                "global_load_async_to_lds_b128 %0, %1, off\n\t"
                "global_load_async_to_lds_b128 %0, %1, off offset:16"
                :: "v"(loff), "v"(ga) : "memory");
        }
    };

    float c[16];
    v16h hB;
#pragma unroll
    for (int s = 0; s < 16; ++s) {
        c[s]  = 0.0f;
        hB[s] = (_Float16)0.0f;
    }

#pragma unroll
    for (int s = 0; s < LA; ++s) issue_stage(s, s);

    for (int t = 0; t < TSZ; ++t) {
        const int slot = t & (NSTAGE - 1);
        if (t + LA < TSZ) {
            issue_stage(t + LA, (t + LA) & (NSTAGE - 1));
            // Async loads retire in order: <=48 outstanding means the 16
            // loads of stage t have landed in LDS.
            asm volatile("s_wait_asynccnt 48" ::: "memory");
        } else {
            asm volatile("s_wait_asynccnt 0" ::: "memory");
        }

        v8f acc[NT];
#pragma unroll
        for (int t8 = 0; t8 < NT; ++t8) {
            v4f lo = *(const v4f*)&xstage[slot][t8][lane][0];
            v4f hi4 = *(const v4f*)&xstage[slot][t8][lane][4];
#pragma unroll
            for (int q = 0; q < 4; ++q) {
                acc[t8][q]     = lo[q];
                acc[t8][4 + q] = hi4[q];
            }
        }

#pragma unroll
        for (int t8 = 0; t8 < NT; ++t8)
            acc[t8] = __builtin_amdgcn_wmma_f32_16x16x32_f16(
                false, A[t8], false, hB, (short)0, acc[t8], false, false);

        // Lane-local gate math. Lane (b = btile*16+m, half hi) owns hidden
        // units u = 16*(s>>3) + 8*hi + (s&7); gates i/f/g/o of u live in
        // tiles {0,1}/{2,3}/{4,5}/{6,7} at register r = s&7.
#pragma unroll
        for (int s = 0; s < 16; ++s) {
            const int p = s >> 3, r = s & 7;
            const float ig = fast_sigmoid(acc[0 + p][r]);
            const float fg = fast_sigmoid(acc[2 + p][r]);
            const float gg = fast_tanh(acc[4 + p][r]);
            const float og = fast_sigmoid(acc[6 + p][r]);
            const float cn = fg * c[s] + ig * gg;
            c[s] = cn;
            // h lands directly in the next step's B-operand slot s.
            hB[s] = (_Float16)(og * fast_tanh(cn));
        }
    }

    // Final FC: out[b, k] = sum_u h[b,u] * fc_w[u,k] + fc_b[k].
#pragma unroll
    for (int k = 0; k < 2; ++k) {
        float part = 0.0f;
#pragma unroll
        for (int s = 0; s < 16; ++s) {
            const int u = 16 * (s >> 3) + 8 * hi + (s & 7);
            part += (float)hB[s] * fc_w[u * 2 + k];
        }
        part += __shfl_xor(part, 16, 32);   // combine the two lane halves
        if (hi == 0)
            out[(btile * 16 + m) * 2 + k] = part + fc_b[k];
    }
}

extern "C" void kernel_launch(void* const* d_in, const int* in_sizes, int n_in,
                              void* d_out, int out_size, void* d_ws, size_t ws_size,
                              hipStream_t stream) {
    (void)in_sizes; (void)n_in; (void)out_size; (void)ws_size;
    const int*   x     = (const int*)d_in[0];
    const float* emb   = (const float*)d_in[1];
    const float* W_ih  = (const float*)d_in[2];
    const float* W_hh  = (const float*)d_in[3];
    const float* bias  = (const float*)d_in[4];
    const float* fc_w  = (const float*)d_in[5];
    const float* fc_b  = (const float*)d_in[6];
    float*       out   = (float*)d_out;
    float*       xg    = (float*)d_ws;   // T*B*4H floats = 256 MiB staging

    dim3 g1(BSZ / 16, TSZ / TCHUNK);     // 16 x 128 blocks, 1 wave each
    lstm_xgate_kernel<<<g1, 32, 0, stream>>>(x, emb, W_ih, bias, xg);

    lstm_scan_kernel<<<BSZ / 16, 32, 0, stream>>>(W_hh, xg, fc_w, fc_b, out);
}
